// DownsampleWithPruning_40312563040280
// MI455X (gfx1250) — matrix operand order repaired and checked by reading, not compiled, not run on hardware
//
#include <hip/hip_runtime.h>
#include <hip/hip_bf16.h>
#include <limits.h>
#include <stdint.h>

// ---------------------------------------------------------------------------
// Sparse stride-2 downsample conv + pruning for MI455X (gfx1250, wave32).
// GEMM path: v_wmma_f32_16x16x32_bf16 (K=32 == CIN, one WMMA per tap/tile).
// Gather path: double-buffered GLOBAL_LOAD_ASYNC_TO_LDS_B128 (ASYNCcnt-
// pipelined) so the next tap's 2KB feature tile streams into LDS while the
// current tap's WMMA executes. Memory-bound: ~1.1GB total -> ~50us @23.3TB/s.
// ---------------------------------------------------------------------------

#define GEXT 256

typedef __attribute__((ext_vector_type(16))) __bf16   v16bf;
typedef __attribute__((ext_vector_type(8)))  float    v8f;
typedef __attribute__((ext_vector_type(8)))  unsigned v8u;

__device__ __forceinline__ unsigned short f2bf(float f) {
  unsigned u = __builtin_bit_cast(unsigned, f);
  u += 0x7fffu + ((u >> 16) & 1u);          // round-to-nearest-even
  return (unsigned short)(u >> 16);
}
__device__ __forceinline__ unsigned pk2bf(float a, float b) {
  return (unsigned)f2bf(a) | ((unsigned)f2bf(b) << 16);
}

// ---------------- init / fill ----------------
__global__ void k_fill_i32(int* p, int v, int n) {
  int i = blockIdx.x * blockDim.x + threadIdx.x;
  if (i < n) p[i] = v;
}
__global__ void k_fill_u64(unsigned long long* p, unsigned long long v, int n) {
  int i = blockIdx.x * blockDim.x + threadIdx.x;
  if (i < n) p[i] = v;
}
__global__ void k_init_scal(int* s) {
  if (threadIdx.x == 0) { s[0] = 0; s[1] = INT_MIN; s[2] = INT_MAX; s[3] = 1; s[4] = 0; }
}

// ---------------- input voxel hash (key -> row) ----------------
__global__ void k_build_in_hash(const int* __restrict__ coords, int n,
                                int* __restrict__ hk, int* __restrict__ hv, int hmask) {
  int i = blockIdx.x * blockDim.x + threadIdx.x;
  if (i >= n) return;
  int b = coords[i*4+0], x = coords[i*4+1], y = coords[i*4+2], z = coords[i*4+3];
  int key = ((b * GEXT + x) * GEXT + y) * GEXT + z;
  unsigned h = ((unsigned)key * 2654435761u) & (unsigned)hmask;
  for (;;) {
    int old = atomicCAS(&hk[h], -1, key);
    if (old == -1 || old == key) { hv[h] = i; break; }
    h = (h + 1) & (unsigned)hmask;
  }
}

// ---------------- unique parent keys (stride-2 downsample) ----------------
__global__ void k_parents(const int* __restrict__ coords, int n,
                          int* __restrict__ oh, int ohmask,
                          int* __restrict__ okey, int* __restrict__ counter) {
  int i = blockIdx.x * blockDim.x + threadIdx.x;
  if (i >= n) return;
  int b = coords[i*4+0], x = coords[i*4+1] & ~1, y = coords[i*4+2] & ~1, z = coords[i*4+3] & ~1;
  int pkey = ((b * GEXT + x) * GEXT + y) * GEXT + z;
  unsigned h = ((unsigned)pkey * 2654435761u) & (unsigned)ohmask;
  for (;;) {
    int old = atomicCAS(&oh[h], -1, pkey);
    if (old == pkey) return;
    if (old == -1) { int pos = atomicAdd(counter, 1); okey[pos] = pkey; return; }
    h = (h + 1) & (unsigned)ohmask;
  }
}

// ---------------- bitonic sort pass (ascending, INT_MAX padding) ----------
__global__ void k_bitonic(int* __restrict__ d, int j, int k, int n) {
  int i = blockIdx.x * blockDim.x + threadIdx.x;
  int ixj = i ^ j;
  if (i < n && ixj > i) {
    int a = d[i], b = d[ixj];
    bool up = ((i & k) == 0);
    if ((a > b) == up) { d[i] = b; d[ixj] = a; }
  }
}

// ---------------- step/shift reductions (faithful to reference) -----------
__global__ void k_minmax_out(const int* __restrict__ okey, int n, int* __restrict__ s) {
  int i = blockIdx.x * blockDim.x + threadIdx.x;
  if (i >= n) return;
  int key = okey[i];
  int mx, mn;
  if (key < 0 || key == INT_MAX) { mx = -1; mn = -1; }     // padded rows are all -1
  else {
    int z = key & 255, y = (key >> 8) & 255, x = (key >> 16) & 255, b = key >> 24;
    mx = max(max(b, x), max(y, z));
    mn = min(min(b, x), min(y, z));
  }
  atomicMax(&s[1], mx); atomicMin(&s[2], mn);
}
__global__ void k_minmax_ref(const int* __restrict__ rc, int n, int* __restrict__ s) {
  int i = blockIdx.x * blockDim.x + threadIdx.x;
  if (i >= n) return;
  int b = rc[i*4+0], x = rc[i*4+1], y = rc[i*4+2], z = rc[i*4+3];
  atomicMax(&s[1], max(max(b, x), max(y, z)));
  atomicMin(&s[2], min(min(b, x), min(y, z)));
}
__global__ void k_finalize(int* s) {
  if (threadIdx.x == 0) { s[3] = s[1] + 1; s[4] = (s[2] < 0) ? -s[2] : 0; }
}

// ---------------- ref-coordinate ravel hash (pruning set) -----------------
__global__ void k_ref_hash(const int* __restrict__ rc, int n,
                           unsigned long long* __restrict__ rh, int rmask,
                           const int* __restrict__ s) {
  int i = blockIdx.x * blockDim.x + threadIdx.x;
  if (i >= n) return;
  long long step = s[3]; int sh = s[4];
  long long c0 = rc[i*4+0];
  long long c1 = rc[i*4+1] + sh, c2 = rc[i*4+2] + sh, c3 = rc[i*4+3] + sh;
  unsigned long long v = (unsigned long long)(c3 + c2*step + c1*step*step + c0*step*step*step);
  unsigned h = ((unsigned)v * 2654435761u) & (unsigned)rmask;
  for (;;) {
    unsigned long long old = atomicCAS(&rh[h], ~0ull, v);
    if (old == ~0ull || old == v) return;
    h = (h + 1) & (unsigned)rmask;
  }
}

__device__ __forceinline__ bool ref_contains(const unsigned long long* __restrict__ rh,
                                             int rmask, unsigned long long v) {
  unsigned h = ((unsigned)v * 2654435761u) & (unsigned)rmask;
  for (int it = 0; it <= rmask; ++it) {
    unsigned long long k = rh[h];
    if (k == v) return true;
    if (k == ~0ull) return false;
    h = (h + 1) & (unsigned)rmask;
  }
  return false;
}

// ---------------- W -> bf16 WMMA B-fragment layout ------------------------
// B (32x16 bf16), wave32: lanes 0-15 hold K=0..15 (col N=lane), lanes 16-31
// hold K=16..31 (col N=lane-16), 2 bf16 per dword, 8 dwords per lane.
__global__ void k_wfrag(const float* __restrict__ W, unsigned* __restrict__ wf) {
  int idx = blockIdx.x * blockDim.x + threadIdx.x;
  if (idx >= 8 * 4 * 32) return;            // tap * ntile * lane
  int l = idx & 31, w = (idx >> 5) & 3, i = idx >> 7;
  int col = 16 * w + (l & 15);
  int kb  = (l < 16) ? 0 : 16;
  for (int t = 0; t < 8; ++t) {
    int K = kb + 2 * t;
    unsigned short u0 = f2bf(W[(i * 32 + K)     * 64 + col]);
    unsigned short u1 = f2bf(W[(i * 32 + K + 1) * 64 + col]);
    wf[idx * 8 + t] = (unsigned)u0 | ((unsigned)u1 << 16);
  }
}

// ---------------- main gather + WMMA GEMM + prune kernel ------------------
__global__ __launch_bounds__(128)
void k_spconv(const float* __restrict__ feats, const unsigned* __restrict__ wfrag,
              const float* __restrict__ bias, const float* __restrict__ zrow,
              const int* __restrict__ okey, int n,
              const int* __restrict__ hk, const int* __restrict__ hv, int hmask,
              const unsigned long long* __restrict__ rh, int rmask,
              const int* __restrict__ scal,
              int* __restrict__ outC, float* __restrict__ outF, float* __restrict__ outM) {
  __shared__ int   sKey[16];
  __shared__ int   sIdx[8][16];
  __shared__ float sMask[16];
  __shared__ __align__(16) float sRaw[2][16][32];   // double-buffered raw f32 A-tiles

  const int tid  = threadIdx.x;
  const int lane = tid & 31;
  const int w    = tid >> 5;                // wave id = N-tile id (COUT/16 = 4)
  const int m0   = blockIdx.x * 16;

  // ---- phase 0: keys, coords output, pruning mask (threads 0..15) ----
  if (tid < 16) {
    int r = tid, row = m0 + r;
    int key = -1; float m = 0.f;
    if (row < n) {
      int k0 = okey[row];
      bool valid = (k0 >= 0 && k0 != INT_MAX);
      int bb = -1, xx = -1, yy = -1, zz = -1;
      if (valid) { key = k0; zz = k0 & 255; yy = (k0 >> 8) & 255; xx = (k0 >> 16) & 255; bb = k0 >> 24; }
      outC[row*4+0] = bb; outC[row*4+1] = xx; outC[row*4+2] = yy; outC[row*4+3] = zz;
      if (valid) {
        long long step = scal[3]; int sh = scal[4];
        unsigned long long v = (unsigned long long)(
            (long long)(zz + sh) + (long long)(yy + sh) * step +
            (long long)(xx + sh) * step * step + (long long)bb * step * step * step);
        if (ref_contains(rh, rmask, v)) m = 1.f;
      }
      outM[row] = m;
    }
    sKey[r] = key; sMask[r] = m;
  }
  __syncthreads();

  // ---- phase 1: one hash lookup per thread (8 taps x 16 rows = 128) ----
  {
    int t = tid >> 4, r = tid & 15;
    int k = sKey[r], idx = -1;
    if (k >= 0) {
      int ck = k + ((t >> 2) & 1) * 65536 + ((t >> 1) & 1) * 256 + (t & 1);
      unsigned h = ((unsigned)ck * 2654435761u) & (unsigned)hmask;
      for (int it = 0; it <= hmask; ++it) {
        int kk = hk[h];
        if (kk == ck) { idx = hv[h]; break; }
        if (kk == -1) break;
        h = (h + 1) & (unsigned)hmask;
      }
    }
    sIdx[t][r] = idx;
    if (idx >= 0) __builtin_prefetch(feats + (size_t)idx * 32, 0, 2);
  }
  __syncthreads();

  // ---- per-wave B fragments for all 8 taps (hoisted out of the loop) ----
  v8u brf[8];
  #pragma unroll
  for (int t = 0; t < 8; ++t)
    brf[t] = *(const v8u*)(wfrag + (((size_t)t * 4 + w) * 32 + lane) * 8);

  const int col = 16 * w + (lane & 15);     // output channel this lane owns
  float bv = bias[col];
  v8f acc = {};

  // ---- async prefill: tap 0 tile -> sRaw[0] (wave 0: 32 lanes x 64B) ----
  if (w == 0) {
    int r = lane & 15, hh = lane >> 4;
    int idx = sIdx[0][r];
    const float* src = (idx >= 0) ? (feats + (size_t)idx * 32 + hh * 16) : (zrow + hh * 16);
    unsigned lp = (unsigned)(uintptr_t)&sRaw[0][r][hh * 16];
    unsigned long long ga = (unsigned long long)(uintptr_t)src;
    asm volatile(
        "global_load_async_to_lds_b128 %0, %1, off\n\t"
        "global_load_async_to_lds_b128 %0, %1, off offset:16\n\t"
        "global_load_async_to_lds_b128 %0, %1, off offset:32\n\t"
        "global_load_async_to_lds_b128 %0, %1, off offset:48"
        :: "v"(lp), "v"(ga) : "memory");
  }

  // ---- pipelined 8-tap loop: issue t+1, wait t, convert + WMMA ----
  #pragma unroll
  for (int t = 0; t < 8; ++t) {
    const int cur = t & 1;
    if (w == 0) {
      if (t < 7) {
        int r = lane & 15, hh = lane >> 4;
        int idx = sIdx[t + 1][r];
        const float* src = (idx >= 0) ? (feats + (size_t)idx * 32 + hh * 16) : (zrow + hh * 16);
        unsigned lp = (unsigned)(uintptr_t)&sRaw[cur ^ 1][r][hh * 16];
        unsigned long long ga = (unsigned long long)(uintptr_t)src;
        asm volatile(
            "global_load_async_to_lds_b128 %0, %1, off\n\t"
            "global_load_async_to_lds_b128 %0, %1, off offset:16\n\t"
            "global_load_async_to_lds_b128 %0, %1, off offset:32\n\t"
            "global_load_async_to_lds_b128 %0, %1, off offset:48"
            :: "v"(lp), "v"(ga) : "memory");
        asm volatile("s_wait_asynccnt 0x4" ::: "memory");   // tap t done, t+1 in flight
      } else {
        asm volatile("s_wait_asynccnt 0x0" ::: "memory");   // last tap fully landed
      }
    }
    __syncthreads();
    // A fragment (contiguous K to match B): lane<16 -> K{0..15},
    // lane>=16 -> K{16..31}; read raw f32 from LDS, pack to bf16.
    int m  = lane & 15;
    int kb = (lane < 16) ? 0 : 16;
    const float4* q0 = (const float4*)&sRaw[cur][m][kb];
    const float4* q1 = (const float4*)&sRaw[cur][m][kb + 8];
    float4 a0 = q0[0], a1 = q0[1], b0 = q1[0], b1 = q1[1];
    v8u ar;
    ar[0] = pk2bf(a0.x, a0.y); ar[1] = pk2bf(a0.z, a0.w);
    ar[2] = pk2bf(a1.x, a1.y); ar[3] = pk2bf(a1.z, a1.w);
    ar[4] = pk2bf(b0.x, b0.y); ar[5] = pk2bf(b0.z, b0.w);
    ar[6] = pk2bf(b1.x, b1.y); ar[7] = pk2bf(b1.z, b1.w);
    v16bf av  = __builtin_bit_cast(v16bf, ar);
    v16bf bvv = __builtin_bit_cast(v16bf, brf[t]);
    acc = __builtin_amdgcn_wmma_f32_16x16x32_bf16(false, av, false, bvv,
                                                  (short)0, acc, false, false);
    __syncthreads();   // protect sRaw[cur] from tap t+2's async writes
  }

  // ---- epilogue: bias + prune, C/D layout: VGPR r holds M = r (+8 hi) ----
  #pragma unroll
  for (int r = 0; r < 8; ++r) {
    int M = r + ((lane < 16) ? 0 : 8);
    int row = m0 + M;
    if (row < n) outF[(size_t)row * 64 + col] = (acc[r] + bv) * sMask[M];
  }
}

// ---------------------------------------------------------------------------
extern "C" void kernel_launch(void* const* d_in, const int* in_sizes, int n_in,
                              void* d_out, int out_size, void* d_ws, size_t ws_size,
                              hipStream_t stream) {
  const float* feats  = (const float*)d_in[0];
  const float* W      = (const float*)d_in[1];
  const float* bias   = (const float*)d_in[2];
  const int*   coords = (const int*)d_in[3];
  const int*   refc   = (const int*)d_in[4];
  const int N    = in_sizes[3] / 4;
  const int NREF = in_sizes[4] / 4;

  auto npow2 = [](int x) { int p = 1; while (p < x) p <<= 1; return p; };
  const int HSZ = npow2(2 * N);     // input / parent hash sizes (load <= 0.5)
  const int PAD = npow2(N);         // bitonic-sorted key array
  const int RSZ = npow2(2 * NREF);  // pruning-set hash

  char* p = (char*)d_ws;
  int* inK  = (int*)p;  p += (size_t)HSZ * 4;
  int* inV  = (int*)p;  p += (size_t)HSZ * 4;
  int* outH = (int*)p;  p += (size_t)HSZ * 4;
  int* okey = (int*)p;  p += (size_t)PAD * 4;
  unsigned long long* refH = (unsigned long long*)(((uintptr_t)p + 7) & ~(uintptr_t)7);
  p = (char*)refH + (size_t)RSZ * 8;
  unsigned* wfrag = (unsigned*)p; p += (size_t)8 * 4 * 32 * 8 * 4;
  int* scal = (int*)p; p += 32;   // [0]=count [1]=gmax [2]=gmin [3]=step [4]=shift
  float* zrow = (float*)(((uintptr_t)p + 127) & ~(uintptr_t)127); // 128B zero page

  const int T = 256;
  k_fill_i32<<<(HSZ + T - 1) / T, T, 0, stream>>>(inK, -1, HSZ);
  k_fill_i32<<<(HSZ + T - 1) / T, T, 0, stream>>>(outH, -1, HSZ);
  k_fill_i32<<<(PAD + T - 1) / T, T, 0, stream>>>(okey, INT_MAX, PAD);
  k_fill_u64<<<(RSZ + T - 1) / T, T, 0, stream>>>(refH, ~0ull, RSZ);
  k_fill_i32<<<1, 32, 0, stream>>>((int*)zrow, 0, 32);
  k_init_scal<<<1, 32, 0, stream>>>(scal);

  k_build_in_hash<<<(N + T - 1) / T, T, 0, stream>>>(coords, N, inK, inV, HSZ - 1);
  k_parents<<<(N + T - 1) / T, T, 0, stream>>>(coords, N, outH, HSZ - 1, okey, scal);

  for (int k = 2; k <= PAD; k <<= 1)
    for (int j = k >> 1; j > 0; j >>= 1)
      k_bitonic<<<(PAD + T - 1) / T, T, 0, stream>>>(okey, j, k, PAD);

  k_minmax_out<<<(N + T - 1) / T, T, 0, stream>>>(okey, N, scal);
  k_minmax_ref<<<(NREF + T - 1) / T, T, 0, stream>>>(refc, NREF, scal);
  k_finalize<<<1, 32, 0, stream>>>(scal);
  k_ref_hash<<<(NREF + T - 1) / T, T, 0, stream>>>(refc, NREF, refH, RSZ - 1, scal);
  k_wfrag<<<(8 * 4 * 32 + T - 1) / T, T, 0, stream>>>(W, wfrag);

  int*   outC = (int*)d_out;
  float* outF = (float*)d_out + (size_t)N * 4;
  float* outM = outF + (size_t)N * 64;
  k_spconv<<<(N + 15) / 16, 128, 0, stream>>>(feats, wfrag, bias, zrow, okey, N,
                                              inK, inV, HSZ - 1, refH, RSZ - 1, scal,
                                              outC, outF, outM);
}
